// MultiModalSurveyGNN_68633577390197
// MI455X (gfx1250) — compile-verified, matrix operand order checked by reading, not run on hardware
//
#include <hip/hip_runtime.h>
#include <hip/hip_bf16.h>

#define NN      16384
#define NBANDS  8
#define HH      256
#define NHEADS  4
#define DH      64
#define KNB     8
#define NLAYERS 3
#define H2      128
#define NOUT    128

typedef __attribute__((ext_vector_type(16))) __bf16 v16bf;
typedef __attribute__((ext_vector_type(8)))  float  v8f;
typedef __attribute__((ext_vector_type(8)))  __bf16 bf16x8;

struct W9 { const float* p[9]; };

// ---------------- utility ----------------
__global__ void zero_f(float* p, int n) {
    int i = blockIdx.x * blockDim.x + threadIdx.x;
    if (i < n) p[i] = 0.0f;
}

// fp32 -> bf16, transposed: WT[ncol][k] = W[k][ncol], 9 matrices of 256x256
__global__ void convert_weights(W9 w, __bf16* out) {
    int id = blockIdx.x * blockDim.x + threadIdx.x;
    if (id >= 9 * HH * HH) return;
    int m = id / (HH * HH);
    int r = id % (HH * HH);
    int ncol = r / HH;
    int k    = r % HH;
    out[(size_t)m * HH * HH + (size_t)ncol * HH + k] =
        (__bf16)(w.p[m][(size_t)k * HH + ncol]);
}

__global__ void sq_kernel(const float* __restrict__ coords, float* __restrict__ sq, int n) {
    int i = blockIdx.x * blockDim.x + threadIdx.x;
    if (i >= n) return;
    float x = coords[3 * i], y = coords[3 * i + 1], z = coords[3 * i + 2];
    sq[i] = x * x + y * y + z * z;
}

// ---------------- stage 1: embeddings (tiny inner dims, VALU) ----------------
__global__ void embed_kernel(const float* __restrict__ phot, const float* __restrict__ coords,
                             const float* __restrict__ Wep, const float* __restrict__ bep,
                             const float* __restrict__ Wea, const float* __restrict__ bea,
                             float* __restrict__ Ep, float* __restrict__ Ea, int n) {
    int id = blockIdx.x * blockDim.x + threadIdx.x;
    if (id >= n * HH) return;
    int i = id >> 8, c = id & 255;
    float sp = bep[c];
#pragma unroll
    for (int j = 0; j < NBANDS; ++j) sp += phot[i * NBANDS + j] * Wep[j * HH + c];
    Ep[id] = sp;
    float sa = bea[c];
#pragma unroll
    for (int j = 0; j < 3; ++j) sa += coords[i * 3 + j] * Wea[j * HH + c];
    Ea[id] = sa;
}

// ---------------- WMMA GEMM: C[M x 256] = A[M x 256](f32) @ W(bf16) + bias ----------------
// WT is the 256x256 weight stored transposed (WT[n][k], bf16).  One wave per
// 16x64 output tile (4 accumulators): each K-step loads one A fragment and
// issues 4 v_wmma_f32_16x16x32_bf16 against 4 B fragments -> 32 WMMA / wave,
// 4x less A traffic than a 16x16-tile mapping.
__global__ void gemm_wmma(const float* __restrict__ A, int lda,
                          const __bf16* __restrict__ WT,
                          const float* __restrict__ bias,
                          float* __restrict__ C, int ldc,
                          int M, int relu) {
    const int lane   = threadIdx.x & 31;
    const int wave   = threadIdx.x >> 5;
    const int tilesN = HH >> 6;                        // 4 blocks of 64 cols
    const int tiles  = (M >> 4) * tilesN;
    int tid = blockIdx.x * (blockDim.x >> 5) + wave;
    if (tid >= tiles) return;
    const int tM  = tid / tilesN;
    const int tN4 = tid - tM * tilesN;                 // which 64-col block
    const int r16 = lane & 15, lg = lane >> 4;
    const float* arow = A + (size_t)(tM * 16 + r16) * lda;
    const __bf16* brow0 = WT + (size_t)(tN4 * 64 + r16) * HH;   // +j*16*HH for j-th 16-col tile

    v8f acc0 = {0.f, 0.f, 0.f, 0.f, 0.f, 0.f, 0.f, 0.f};
    v8f acc1 = acc0, acc2 = acc0, acc3 = acc0;

#pragma unroll
    for (int kk = 0; kk < HH; kk += 32) {
        const int k0 = kk + 8 * lg;       // elements e=0..7  -> K = k0..k0+7
        const int k1 = k0 + 16;           // elements e=8..15 -> K = k1..k1+7
        float at[16];
        *(float4*)(at + 0)  = *(const float4*)(arow + k0);
        *(float4*)(at + 4)  = *(const float4*)(arow + k0 + 4);
        *(float4*)(at + 8)  = *(const float4*)(arow + k1);
        *(float4*)(at + 12) = *(const float4*)(arow + k1 + 4);
        v16bf a;
#pragma unroll
        for (int e = 0; e < 8; ++e) {
            a[e]     = (__bf16)at[e];
            a[8 + e] = (__bf16)at[8 + e];
        }
#pragma unroll
        for (int j = 0; j < 4; ++j) {
            const __bf16* brow = brow0 + (size_t)j * 16 * HH;
            bf16x8 b0 = *(const bf16x8*)(brow + k0);
            bf16x8 b1 = *(const bf16x8*)(brow + k1);
            v16bf b;
#pragma unroll
            for (int e = 0; e < 8; ++e) { b[e] = b0[e]; b[8 + e] = b1[e]; }
            v8f cin = (j == 0) ? acc0 : (j == 1) ? acc1 : (j == 2) ? acc2 : acc3;
            v8f cout = __builtin_amdgcn_wmma_f32_16x16x32_bf16(
                /*neg_a=*/false, a, /*neg_b=*/false, b,
                /*c_mod=*/(short)0, cin, /*reuse_a=*/false, /*reuse_b=*/false);
            if (j == 0) acc0 = cout; else if (j == 1) acc1 = cout;
            else if (j == 2) acc2 = cout; else acc3 = cout;
        }
    }
#pragma unroll
    for (int j = 0; j < 4; ++j) {
        v8f acc = (j == 0) ? acc0 : (j == 1) ? acc1 : (j == 2) ? acc2 : acc3;
#pragma unroll
        for (int r = 0; r < 8; ++r) {                 // C/D layout: m = 8*lg + r, n = r16
            int m = tM * 16 + 8 * lg + r;
            int n = tN4 * 64 + j * 16 + r16;
            float v = acc[r] + bias[n];
            if (relu) v = fmaxf(v, 0.0f);
            C[(size_t)m * ldc + n] = v;
        }
    }
}

// ---------------- stage 2: 2-token / 4-head attention (scalar, tiny) ----------------
__global__ void attention_kernel(const float* __restrict__ q, const float* __restrict__ k,
                                 const float* __restrict__ v, float* __restrict__ o, int n) {
    int id = blockIdx.x * blockDim.x + threadIdx.x;
    if (id >= n * NHEADS) return;
    int node = id >> 2, h = id & 3;
    size_t b0 = (size_t)(2 * node) * HH + h * DH;
    size_t b1 = b0 + HH;
    const float *q0 = q + b0, *q1 = q + b1, *k0 = k + b0, *k1 = k + b1;
    const float *v0 = v + b0, *v1 = v + b1;
    float s00 = 0, s01 = 0, s10 = 0, s11 = 0;
#pragma unroll 8
    for (int d = 0; d < DH; ++d) {
        float a0 = q0[d], a1 = q1[d], c0 = k0[d], c1 = k1[d];
        s00 += a0 * c0; s01 += a0 * c1; s10 += a1 * c0; s11 += a1 * c1;
    }
    const float sc = 0.125f;  // 1/sqrt(64)
    s00 *= sc; s01 *= sc; s10 *= sc; s11 *= sc;
    float m0 = fmaxf(s00, s01), m1 = fmaxf(s10, s11);
    float e00 = __expf(s00 - m0), e01 = __expf(s01 - m0);
    float e10 = __expf(s10 - m1), e11 = __expf(s11 - m1);
    float r0 = 1.0f / (e00 + e01), r1 = 1.0f / (e10 + e11);
    float a00 = e00 * r0, a01 = e01 * r0, a10 = e10 * r1, a11 = e11 * r1;
    float *o0 = o + b0, *o1 = o + b1;
#pragma unroll 8
    for (int d = 0; d < DH; ++d) {
        float x0 = v0[d], x1 = v1[d];
        o0[d] = a00 * x0 + a01 * x1;
        o1[d] = a10 * x0 + a11 * x1;
    }
}

__global__ void fuse_kernel(const float* __restrict__ proj, float* __restrict__ fused, int n) {
    int id = blockIdx.x * blockDim.x + threadIdx.x;
    if (id >= n * HH) return;
    int i = id >> 8, c = id & 255;
    fused[id] = 0.5f * (proj[(size_t)(2 * i) * HH + c] + proj[(size_t)(2 * i + 1) * HH + c]);
}

// ---------------- stage 3: brute-force KNN (LDS-tiled, register top-9) ----------------
__global__ void knn_kernel(const float* __restrict__ coords, const float* __restrict__ sq,
                           int* __restrict__ nbr, int n) {
    __shared__ float sx[256], sy[256], sz[256], ss[256];
    int i = blockIdx.x * 256 + threadIdx.x;
    float xi = coords[3 * i], yi = coords[3 * i + 1], zi = coords[3 * i + 2], qi = sq[i];
    float bd[KNB + 1];
    int   bi[KNB + 1];
#pragma unroll
    for (int t = 0; t <= KNB; ++t) { bd[t] = 3.0e38f; bi[t] = 0; }
    for (int jt = 0; jt < n; jt += 256) {
        __syncthreads();
        int j = jt + threadIdx.x;
        sx[threadIdx.x] = coords[3 * j];
        sy[threadIdx.x] = coords[3 * j + 1];
        sz[threadIdx.x] = coords[3 * j + 2];
        ss[threadIdx.x] = sq[j];
        __syncthreads();
        for (int t = 0; t < 256; ++t) {
            float d2 = qi + ss[t] - 2.0f * (xi * sx[t] + yi * sy[t] + zi * sz[t]);
            if (d2 < bd[KNB]) {
                int j2 = jt + t;
                int p = KNB;
                while (p > 0 && d2 < bd[p - 1]) { bd[p] = bd[p - 1]; bi[p] = bi[p - 1]; --p; }
                bd[p] = d2; bi[p] = j2;
            }
        }
    }
#pragma unroll
    for (int t = 0; t < KNB; ++t) nbr[(size_t)i * KNB + t] = bi[t + 1];  // drop self (rank 0)
}

// ---------------- stage 4: graph degree + scatter ----------------
__global__ void deg_count_kernel(const int* __restrict__ nbr, float* __restrict__ deg, int m) {
    int id = blockIdx.x * blockDim.x + threadIdx.x;
    if (id >= m) return;
    atomicAdd(&deg[nbr[id]], 1.0f);
}

__global__ void deg_fin_kernel(float* __restrict__ deg, float* __restrict__ inv, int n) {
    int id = blockIdx.x * blockDim.x + threadIdx.x;
    if (id >= n) return;
    float d = deg[id] + 1.0f;
    deg[id] = d;
    inv[id] = rsqrtf(d);
}

__global__ void scatter_kernel(const float* __restrict__ h, const int* __restrict__ nbr,
                               const float* __restrict__ inv, float* __restrict__ agg, int n) {
    long id = (long)blockIdx.x * blockDim.x + threadIdx.x;
    if (id >= (long)n * KNB * HH) return;
    int c = (int)(id & 255);
    int e = (int)(id >> 8);
    int i = e >> 3;           // src
    int dst = nbr[e];
    atomicAdd(&agg[(size_t)dst * HH + c], h[(size_t)i * HH + c] * inv[i] * inv[dst]);
}

__global__ void selfadd_kernel(const float* __restrict__ h, const float* __restrict__ deg,
                               float* __restrict__ agg, int n) {
    int id = blockIdx.x * blockDim.x + threadIdx.x;
    if (id >= n * HH) return;
    int i = id >> 8;
    agg[id] += h[id] / deg[i];
}

// ---------------- stage 5: pool + MLP head ----------------
__global__ void pool_kernel(const float* __restrict__ h, float* __restrict__ g, int n) {
    int c = threadIdx.x;   // single block of 256
    float s = 0.0f;
    for (int i = 0; i < n; ++i) s += h[(size_t)i * HH + c];
    g[c] = s / (float)n;
}

__global__ void mlp_kernel(const float* __restrict__ g,
                           const float* __restrict__ W1, const float* __restrict__ b1,
                           const float* __restrict__ W2, const float* __restrict__ b2,
                           float* __restrict__ out) {
    __shared__ float sg[HH];
    __shared__ float st[H2];
    int t = threadIdx.x;   // single block of 128
    sg[t] = g[t];
    sg[t + 128] = g[t + 128];
    __syncthreads();
    float acc = b1[t];
    for (int c = 0; c < HH; ++c) acc += sg[c] * W1[c * H2 + t];
    st[t] = fmaxf(acc, 0.0f);
    __syncthreads();
    float o = b2[t];
    for (int j = 0; j < H2; ++j) o += st[j] * W2[j * NOUT + t];
    out[t] = o;
}

// ---------------- launch ----------------
extern "C" void kernel_launch(void* const* d_in, const int* in_sizes, int n_in,
                              void* d_out, int out_size, void* d_ws, size_t ws_size,
                              hipStream_t stream) {
    (void)in_sizes; (void)n_in; (void)out_size; (void)ws_size;
    const float* phot     = (const float*)d_in[0];
    const float* coords   = (const float*)d_in[1];
    const float* We_phot  = (const float*)d_in[2];
    const float* be_phot  = (const float*)d_in[3];
    const float* We_astro = (const float*)d_in[4];
    const float* be_astro = (const float*)d_in[5];
    const float* Wp_phot  = (const float*)d_in[6];
    const float* bp_phot  = (const float*)d_in[7];
    const float* Wp_astro = (const float*)d_in[8];
    const float* bp_astro = (const float*)d_in[9];
    const float* Wq = (const float*)d_in[10]; const float* bq = (const float*)d_in[11];
    const float* Wk = (const float*)d_in[12]; const float* bk = (const float*)d_in[13];
    const float* Wv = (const float*)d_in[14]; const float* bv = (const float*)d_in[15];
    const float* Wo = (const float*)d_in[16]; const float* bo = (const float*)d_in[17];
    const float* gcn_W = (const float*)d_in[18];
    const float* gcn_b = (const float*)d_in[19];
    const float* W1 = (const float*)d_in[20]; const float* b1 = (const float*)d_in[21];
    const float* W2 = (const float*)d_in[22]; const float* b2 = (const float*)d_in[23];

    char* ws = (char*)d_ws;
    size_t off = 0;
    auto take = [&](size_t bytes) -> char* {
        char* p = ws + off;
        off += (bytes + 255) & ~(size_t)255;
        return p;
    };
    __bf16* wbf  = (__bf16*)take((size_t)9 * HH * HH * sizeof(__bf16));
    float*  bufA = (float*)take((size_t)2 * NN * HH * 4);  // x -> o
    float*  bufB = (float*)take((size_t)2 * NN * HH * 4);  // q -> proj
    float*  bufC = (float*)take((size_t)2 * NN * HH * 4);  // k
    float*  bufD = (float*)take((size_t)2 * NN * HH * 4);  // v
    float*  bufE = (float*)take((size_t)NN * HH * 4);      // E_phot -> fused/h
    float*  bufF = (float*)take((size_t)NN * HH * 4);      // E_astro -> agg
    int*    nbr  = (int*)  take((size_t)NN * KNB * 4);
    float*  deg  = (float*)take((size_t)NN * 4);
    float*  inv  = (float*)take((size_t)NN * 4);
    float*  sq   = (float*)take((size_t)NN * 4);
    float*  g    = (float*)take((size_t)HH * 4);

    W9 w9;
    w9.p[0] = Wp_phot; w9.p[1] = Wp_astro; w9.p[2] = Wq; w9.p[3] = Wk; w9.p[4] = Wv;
    w9.p[5] = Wo; w9.p[6] = gcn_W; w9.p[7] = gcn_W + HH * HH; w9.p[8] = gcn_W + 2 * HH * HH;

    convert_weights<<<(9 * HH * HH + 255) / 256, 256, 0, stream>>>(w9, wbf);
    sq_kernel<<<NN / 256, 256, 0, stream>>>(coords, sq, NN);
    embed_kernel<<<NN * HH / 256, 256, 0, stream>>>(phot, coords, We_phot, be_phot,
                                                    We_astro, be_astro, bufE, bufF, NN);

    // one wave per 16x64 tile; 8 waves per block
    const int tilesN  = (NN / 16) * (HH / 64);          // 4096 tiles
    const int tiles2N = (2 * NN / 16) * (HH / 64);      // 8192 tiles

    // pf/af projections -> interleaved x (row 2n = phot token, row 2n+1 = astro token)
    gemm_wmma<<<tilesN / 8, 256, 0, stream>>>(bufE, HH, wbf + 0 * HH * HH, bp_phot,
                                              bufA, 2 * HH, NN, 0);
    gemm_wmma<<<tilesN / 8, 256, 0, stream>>>(bufF, HH, wbf + 1 * HH * HH, bp_astro,
                                              bufA + HH, 2 * HH, NN, 0);
    // q / k / v projections on (2N x H)
    gemm_wmma<<<tiles2N / 8, 256, 0, stream>>>(bufA, HH, wbf + 2 * HH * HH, bq, bufB, HH, 2 * NN, 0);
    gemm_wmma<<<tiles2N / 8, 256, 0, stream>>>(bufA, HH, wbf + 3 * HH * HH, bk, bufC, HH, 2 * NN, 0);
    gemm_wmma<<<tiles2N / 8, 256, 0, stream>>>(bufA, HH, wbf + 4 * HH * HH, bv, bufD, HH, 2 * NN, 0);

    attention_kernel<<<NN * NHEADS / 256, 256, 0, stream>>>(bufB, bufC, bufD, bufA, NN);
    gemm_wmma<<<tiles2N / 8, 256, 0, stream>>>(bufA, HH, wbf + 5 * HH * HH, bo, bufB, HH, 2 * NN, 0);
    fuse_kernel<<<NN * HH / 256, 256, 0, stream>>>(bufB, bufE, NN);

    knn_kernel<<<NN / 256, 256, 0, stream>>>(coords, sq, nbr, NN);
    zero_f<<<NN / 256, 256, 0, stream>>>(deg, NN);
    deg_count_kernel<<<NN * KNB / 256, 256, 0, stream>>>(nbr, deg, NN * KNB);
    deg_fin_kernel<<<NN / 256, 256, 0, stream>>>(deg, inv, NN);

    for (int l = 0; l < NLAYERS; ++l) {
        zero_f<<<NN * HH / 256, 256, 0, stream>>>(bufF, NN * HH);
        scatter_kernel<<<(NN * KNB * HH) / 256, 256, 0, stream>>>(bufE, nbr, inv, bufF, NN);
        selfadd_kernel<<<NN * HH / 256, 256, 0, stream>>>(bufE, deg, bufF, NN);
        gemm_wmma<<<tilesN / 8, 256, 0, stream>>>(bufF, HH, wbf + (6 + l) * HH * HH,
                                                  gcn_b + l * HH, bufE, HH, NN, 1);
    }

    pool_kernel<<<1, HH, 0, stream>>>(bufE, g, NN);
    mlp_kernel<<<1, H2, 0, stream>>>(g, W1, b1, W2, b2, (float*)d_out);
}
